// Attention_19722489823916
// MI455X (gfx1250) — matrix operand order repaired and checked by reading, not compiled, not run on hardware
//
#include <hip/hip_runtime.h>
#include <hip/hip_bf16.h>
#include <math.h>

// ---------------------------------------------------------------------------
// Triangle attention for MI455X (gfx1250): wave32 + v_wmma_f32_16x16x32_f16.
// Memory-bound on streaming the 268MB f32 triangle_bias once (~11.5us floor
// at 23.3 TB/s). Bias is streamed with a 4-deep async-copy pipeline
// (global_load_async_to_lds_b128 / s_wait_asynccnt); all other intermediates
// are f16 (except the f32 gate G) and live in L2 (192MB).
// ---------------------------------------------------------------------------

typedef _Float16 v16h __attribute__((ext_vector_type(16)));
typedef _Float16 h8   __attribute__((ext_vector_type(8)));
typedef float    v8f  __attribute__((ext_vector_type(8)));
typedef float    f4   __attribute__((ext_vector_type(4)));

#define WMMA(a, b, c) \
  __builtin_amdgcn_wmma_f32_16x16x32_f16(false, (a), false, (b), (short)0, (c), false, false)

static __device__ __forceinline__ v16h pack16(h8 lo, h8 hi) {
  v16h r;
#pragma unroll
  for (int i = 0; i < 8; ++i) { r[i] = lo[i]; r[i + 8] = hi[i]; }
  return r;
}

// Issue one 16x32-f32 bias tile (2KB) as 4 coalesced async b128 copies.
// gbase points at bias[q0][kb]; row stride 2048 floats. LDS layout: row-major
// 16x32 f32 (row stride 128B), chunk c (16B) at ldsoff + c*16.
static __device__ __forceinline__ void bias_async_tile(const float* gbase,
                                                       unsigned ldsoff, int lane) {
#pragma unroll
  for (int i = 0; i < 4; ++i) {
    int chunk = i * 32 + lane;          // 0..127
    int row = chunk >> 3;               // 8 chunks (128B) per row
    int c4 = (chunk & 7) * 4;           // float column
    const float* ga = gbase + (size_t)row * 2048 + c4;
    unsigned la = ldsoff + (unsigned)chunk * 16;
    asm volatile("global_load_async_to_lds_b128 %0, %1, off"
                 :: "v"(la), "v"(ga) : "memory");
  }
}

// ---------------------------------------------------------------------------
// Kernel 1: transpose + f32->f16 convert the five 256x256 weights.
// Output layout wT[sel][n*256 + c] = w[c*256 + n]  (B-fragment contiguous).
// ---------------------------------------------------------------------------
__global__ void prep_weights(const float* __restrict__ wq, const float* __restrict__ wk,
                             const float* __restrict__ wv, const float* __restrict__ wg,
                             const float* __restrict__ wo, _Float16* __restrict__ wT) {
  int idx = blockIdx.x * blockDim.x + threadIdx.x;  // 0..65535
  int n = idx >> 8, c = idx & 255;
  const float* srcs[5] = {wq, wk, wv, wg, wo};
#pragma unroll
  for (int s = 0; s < 5; ++s)
    wT[s * 65536 + n * 256 + c] = (_Float16)srcs[s][c * 256 + n];
}

// ---------------------------------------------------------------------------
// Kernel 2: projections. sel 0:Q(scaled) 1:K 2:V(transposed) 3:gate(sigmoid).
// 4 waves/block, each wave owns a 16-row x 16-col tile, K-loop 256 in steps 32.
// ---------------------------------------------------------------------------
__global__ __launch_bounds__(128) void proj_kernel(
    const float* __restrict__ qx, const float* __restrict__ kvx,
    const float* __restrict__ bg, const _Float16* __restrict__ wT,
    _Float16* __restrict__ Qh, _Float16* __restrict__ Kh,
    _Float16* __restrict__ Vt, float* __restrict__ G) {
  const int sel = blockIdx.z & 3, b = blockIdx.z >> 2;
  const int wave = threadIdx.x >> 5, lane = threadIdx.x & 31;
  const int g = lane >> 4, n = lane & 15, am = lane & 15;
  const int row0 = blockIdx.x * 64 + wave * 16;
  const int n0 = blockIdx.y * 16;

  const float* x = (sel == 1 || sel == 2) ? kvx : qx;
  const _Float16* w = wT + sel * 65536;
  const float* xrow = x + (size_t)(b * 2048 + row0 + am) * 256;

  v8f acc = {};
#pragma unroll
  for (int k0 = 0; k0 < 256; k0 += 32) {
    // A fragment: 16 contiguous cols k0+g*16..k0+g*16+15
    f4 xa0 = *(const f4*)(xrow + k0 + g * 16);
    f4 xa1 = *(const f4*)(xrow + k0 + g * 16 + 4);
    f4 xb0 = *(const f4*)(xrow + k0 + g * 16 + 8);
    f4 xb1 = *(const f4*)(xrow + k0 + g * 16 + 12);
    v16h a;
#pragma unroll
    for (int i = 0; i < 4; ++i) {
      a[i]      = (_Float16)xa0[i];
      a[4 + i]  = (_Float16)xa1[i];
      a[8 + i]  = (_Float16)xb0[i];
      a[12 + i] = (_Float16)xb1[i];
    }
    // B fragment: contiguous 16 halves from transposed weight row.
    const _Float16* wp = w + (n0 + n) * 256 + k0 + g * 16;
    v16h bm = pack16(*(const h8*)wp, *(const h8*)(wp + 8));
    acc = WMMA(a, bm, acc);
  }

  const int col = n0 + n, h = col >> 5, ch = col & 31;
#pragma unroll
  for (int r = 0; r < 8; ++r) {
    int row = row0 + r + 8 * g;  // C fragment row mapping
    float v = acc[r];
    if (sel == 0) {
      Qh[((size_t)(b * 8 + h) * 2048 + row) * 32 + ch] =
          (_Float16)(v * 0.17677669529663687f);  // 1/sqrt(32)
    } else if (sel == 1) {
      Kh[((size_t)(b * 8 + h) * 2048 + row) * 32 + ch] = (_Float16)v;
    } else if (sel == 2) {
      Vt[((size_t)(b * 8 + h) * 32 + ch) * 2048 + row] = (_Float16)v;
    } else {
      float z = v + bg[col];
      G[((size_t)(b * 2048) + row) * 256 + col] = 1.0f / (1.0f + __expf(-z));
    }
  }
}

// ---------------------------------------------------------------------------
// Kernel 3: flash attention. One wave = 16 q rows of one (b,h). Online softmax
// over K in 32-wide steps: 2 score WMMAs + 2 P.V WMMAs per step. The f32 bias
// tile is streamed through a 4-deep async LDS pipeline (up to 3 tiles in
// flight). All LDS is wave-private so no block barriers are needed (per-wave
// LDS ops are in order). Row-sum reduction is deferred to the epilogue: only
// the row max (needed for P consistency) is butterflied per step.
// ---------------------------------------------------------------------------
__global__ __launch_bounds__(128) void attn_kernel(
    const float* __restrict__ mask, const float* __restrict__ tb,
    const _Float16* __restrict__ Qh, const _Float16* __restrict__ Kh,
    const _Float16* __restrict__ Vt, const float* __restrict__ G,
    _Float16* __restrict__ Og) {
  __shared__ __align__(16) float biasLds[4][4][16 * 32];   // 32KB: per-wave 4-deep ring
  __shared__ __align__(16) _Float16 plds[4][16 * 40];      // P staging, 80B row stride

  const int wave = threadIdx.x >> 5, lane = threadIdx.x & 31;
  const int g = lane >> 4, n = lane & 15, am = lane & 15;
  const int q0 = blockIdx.x * 64 + wave * 16;
  const int h = blockIdx.y, b = blockIdx.z;
  const size_t bh = (size_t)(b * 8 + h);

  // Q tile A fragment (loaded once, already pre-scaled).
  const _Float16* qrow = Qh + (bh * 2048 + q0 + am) * 32;
  const v16h aq = pack16(*(const h8*)(qrow + g * 16), *(const h8*)(qrow + g * 16 + 8));

  v8f o0 = {}, o1 = {};
  float mrow[8], lrow[8];
#pragma unroll
  for (int r = 0; r < 8; ++r) { mrow[r] = -1e30f; lrow[r] = 0.0f; }

  const float* tbq = tb + bh * 2048 * 2048 + (size_t)q0 * 2048;  // bias[q0][0]
  const float* mk = mask + (size_t)b * 2048;
  _Float16* lw = plds[wave];
  const unsigned boff = (unsigned)(uintptr_t)&biasLds[wave][0][0];

  // Prologue: start async copies of the first 3 bias tiles.
#pragma unroll
  for (int t = 0; t < 3; ++t)
    bias_async_tile(tbq + t * 32, boff + (unsigned)t * 2048u, lane);

  for (int kb = 0, tile = 0; kb < 2048; kb += 32, ++tile) {
    // Keep 3 tiles in flight, then wait until the current tile has landed.
    // Async loads retire in order; 4 ops per tile -> asynccnt thresholds.
    if (kb + 96 < 2048)
      bias_async_tile(tbq + kb + 96, boff + (unsigned)((tile + 3) & 3) * 2048u, lane);
    int rem = (2048 - 32 - kb) >> 5;  // tiles newer than current still pending
    if (rem >= 3)      asm volatile("s_wait_asynccnt 12" ::: "memory");
    else if (rem == 2) asm volatile("s_wait_asynccnt 8"  ::: "memory");
    else if (rem == 1) asm volatile("s_wait_asynccnt 4"  ::: "memory");
    else               asm volatile("s_wait_asynccnt 0"  ::: "memory");

    // K^T B fragments: row (kb+n) of Kh, channels g*16..g*16+15 contiguous.
    const _Float16* k0p = Kh + (bh * 2048 + kb + n) * 32 + g * 16;
    const _Float16* k1p = Kh + (bh * 2048 + kb + 16 + n) * 32 + g * 16;
    v16h bk0 = pack16(*(const h8*)k0p, *(const h8*)(k0p + 8));
    v16h bk1 = pack16(*(const h8*)k1p, *(const h8*)(k1p + 8));
    v8f s0 = {}, s1 = {};
    s0 = WMMA(aq, bk0, s0);
    s1 = WMMA(aq, bk1, s1);

    // Bias (from LDS tile) + mask add (C fragment: row r+8*g, col kb+n).
    const float* bb = &biasLds[wave][tile & 3][0];
    float m0 = mk[kb + n], m1 = mk[kb + 16 + n];
#pragma unroll
    for (int r = 0; r < 8; ++r) {
      int brow = (r + 8 * g) * 32;
      s0[r] += bb[brow + n] + m0;
      s1[r] += bb[brow + 16 + n] + m1;
    }

    // Online softmax. Max butterfly (xor 1/2/4/8) stays within each 16-lane
    // half = the C-fragment row split on wave32. Row-sum stays lane-partial.
    float p0[8], p1[8];
#pragma unroll
    for (int r = 0; r < 8; ++r) {
      float mr = fmaxf(s0[r], s1[r]);
      mr = fmaxf(mr, __shfl_xor(mr, 1, 32));
      mr = fmaxf(mr, __shfl_xor(mr, 2, 32));
      mr = fmaxf(mr, __shfl_xor(mr, 4, 32));
      mr = fmaxf(mr, __shfl_xor(mr, 8, 32));
      float nm = fmaxf(mrow[r], mr);
      float sc = __expf(mrow[r] - nm);   // row-uniform rescale
      mrow[r] = nm;
      p0[r] = __expf(s0[r] - nm);
      p1[r] = __expf(s1[r] - nm);
      lrow[r] = lrow[r] * sc + (p0[r] + p1[r]);  // per-lane partial sum
      o0[r] *= sc;
      o1[r] *= sc;
    }

    // Re-layout P: C fragment -> wave-private LDS -> A fragment.
    // Same-wave LDS ops are in order; compiler inserts the dscnt wait.
#pragma unroll
    for (int r = 0; r < 8; ++r) {
      int row = r + 8 * g;
      lw[row * 40 + n] = (_Float16)p0[r];
      lw[row * 40 + 16 + n] = (_Float16)p1[r];
    }
    v16h ap = pack16(*(const h8*)&lw[am * 40 + g * 16],
                     *(const h8*)&lw[am * 40 + g * 16 + 8]);

    // V B fragments from transposed V: contiguous 16 halves along k.
    const _Float16* v0p = Vt + (bh * 32 + n) * 2048 + kb + g * 16;
    const _Float16* v1p = Vt + (bh * 32 + 16 + n) * 2048 + kb + g * 16;
    v16h bv0 = pack16(*(const h8*)v0p, *(const h8*)(v0p + 8));
    v16h bv1 = pack16(*(const h8*)v1p, *(const h8*)(v1p + 8));
    o0 = WMMA(ap, bv0, o0);
    o1 = WMMA(ap, bv1, o1);
  }

  // Epilogue: complete the deferred row-sum reduction, normalize, gate, store.
#pragma unroll
  for (int r = 0; r < 8; ++r) {
    float l = lrow[r];
    l += __shfl_xor(l, 1, 32);
    l += __shfl_xor(l, 2, 32);
    l += __shfl_xor(l, 4, 32);
    l += __shfl_xor(l, 8, 32);
    float inv = 1.0f / l;
    int row = q0 + r + 8 * g;
    size_t gbase = ((size_t)(b * 2048) + row) * 256 + h * 32;
    float g0 = G[gbase + n], g1 = G[gbase + 16 + n];
    Og[gbase + n] = (_Float16)(o0[r] * inv * g0);
    Og[gbase + 16 + n] = (_Float16)(o1[r] * inv * g1);
  }
}

// ---------------------------------------------------------------------------
// Kernel 4: out = Og @ w_o + b_o   ([4096,256] x [256,256], f16 WMMA, f32 out)
// ---------------------------------------------------------------------------
__global__ __launch_bounds__(128) void out_gemm(
    const _Float16* __restrict__ Og, const _Float16* __restrict__ woT,
    const float* __restrict__ bo, float* __restrict__ out) {
  const int wave = threadIdx.x >> 5, lane = threadIdx.x & 31;
  const int g = lane >> 4, n = lane & 15, am = lane & 15;
  const int row0 = blockIdx.x * 64 + wave * 16;
  const int n0 = blockIdx.y * 16;

  const _Float16* arow = Og + (size_t)(row0 + am) * 256;
  v8f acc = {};
#pragma unroll
  for (int k0 = 0; k0 < 256; k0 += 32) {
    v16h a = pack16(*(const h8*)(arow + k0 + g * 16),
                    *(const h8*)(arow + k0 + g * 16 + 8));
    const _Float16* wp = woT + (n0 + n) * 256 + k0 + g * 16;
    v16h bm = pack16(*(const h8*)wp, *(const h8*)(wp + 8));
    acc = WMMA(a, bm, acc);
  }
  float bias = bo[n0 + n];
#pragma unroll
  for (int r = 0; r < 8; ++r) {
    int row = row0 + r + 8 * g;
    out[(size_t)row * 256 + n0 + n] = acc[r] + bias;
  }
}

// ---------------------------------------------------------------------------
// Launch. Workspace layout (bytes):
//   [0, 640K)          : 5 transposed f16 weights (wq,wk,wv,wg,wo)
//   [640K, +2M)        : Qh  f16 [B,H,Q,CH]  (pre-scaled)
//   [.., +2M)          : Kh  f16 [B,H,K,CH]
//   [.., +2M)          : Vt  f16 [B,H,CH,K]
//   [.., +4M)          : G   f32 [B,Q,256]   (sigmoid gate)
//   [.., +2M)          : Og  f16 [B,Q,256]   (gated attention output)
// Total ~12.6 MB.
// ---------------------------------------------------------------------------
extern "C" void kernel_launch(void* const* d_in, const int* in_sizes, int n_in,
                              void* d_out, int out_size, void* d_ws, size_t ws_size,
                              hipStream_t stream) {
  const float* qx   = (const float*)d_in[0];
  const float* kvx  = (const float*)d_in[1];
  const float* mask = (const float*)d_in[2];
  const float* tb   = (const float*)d_in[3];
  const float* wq   = (const float*)d_in[4];
  const float* wk   = (const float*)d_in[5];
  const float* wv   = (const float*)d_in[6];
  const float* wg   = (const float*)d_in[7];
  const float* bg   = (const float*)d_in[8];
  const float* wo   = (const float*)d_in[9];
  const float* bo   = (const float*)d_in[10];

  char* ws = (char*)d_ws;
  _Float16* wT = (_Float16*)ws;                                     // 640 KB
  _Float16* Qh = (_Float16*)(ws + 655360);                          // 2 MB
  _Float16* Kh = (_Float16*)(ws + 655360 + 2097152);                // 2 MB
  _Float16* Vt = (_Float16*)(ws + 655360 + 2 * 2097152);            // 2 MB
  float*    G  = (float*)   (ws + 655360 + 3 * 2097152);            // 4 MB
  _Float16* Og = (_Float16*)(ws + 655360 + 3 * 2097152 + 4194304);  // 2 MB
  float* out = (float*)d_out;

  prep_weights<<<256, 256, 0, stream>>>(wq, wk, wv, wg, wo, wT);

  dim3 pg(32, 16, 8);  // 2048/64 row tiles, 256/16 col tiles, B*4 matrices
  proj_kernel<<<pg, 128, 0, stream>>>(qx, kvx, bg, wT, Qh, Kh, Vt, G);

  dim3 ag(32, 8, 2);   // Q/64, H, B
  attn_kernel<<<ag, 128, 0, stream>>>(mask, tb, Qh, Kh, Vt, G, Og);

  dim3 og(64, 16, 1);  // (B*Q)/64, 256/16
  out_gemm<<<og, 128, 0, stream>>>(Og, wT + 4 * 65536, bo, out);
}
